// SwinTransformerBlock_12541304504624
// MI455X (gfx1250) — matrix operand order repaired and checked by reading, not compiled, not run on hardware
//
#include <hip/hip_runtime.h>
#include <hip/hip_bf16.h>
#include <cmath>

// ---------------------------------------------------------------------------
// Swin block for MI455X (gfx1250, wave32, WMMA + TDM).
// ---------------------------------------------------------------------------

typedef __attribute__((ext_vector_type(16))) _Float16 v16h;
typedef __attribute__((ext_vector_type(8)))  _Float16 v8h;
typedef __attribute__((ext_vector_type(8)))  float    v8f;
typedef __attribute__((ext_vector_type(4)))  unsigned int v4u;
typedef __attribute__((ext_vector_type(8)))  int       v8i;
typedef __attribute__((ext_vector_type(4)))  int       v4i;

#define DIM   192
#define NH    6
#define HD    32
#define WS_   7
#define NTOK  49          // tokens per window
#define NWIN  64          // windows per image
#define HH    56
#define MTOT  200704      // B * H * W tokens
#define MLPD  768
#define QSCALE 0.17677669529663687f

// ---------------- WMMA fragment helpers (ISA 7.12.2 layouts) ----------------
// A 16x32 f16: lanes 0-15 row M=lane, K 0..7 & 16..23; lanes 16-31 same row,
// K 8..15 & 24..31.  LDS tile is row-major [m][k], k-contiguous.
__device__ inline v16h frag_a_ld(const _Float16* s, int m0, int ld) {
  const int lane  = threadIdx.x & 31;
  const _Float16* p = s + (size_t)(m0 + (lane & 15)) * ld + ((lane >> 4) << 3);
  v8h lo = *(const v8h*)p;
  v8h hi = *(const v8h*)(p + 16);
  v16h f;
#pragma unroll
  for (int i = 0; i < 8; ++i) { f[i] = lo[i]; f[8 + i] = hi[i]; }
  return f;
}

// B 32x16 f16: lanes 0-15 col N=lane, K 0..7 & 16..23; lanes 16-31 K 8..15 & 24..31.
// LDS tile stored transposed: row = output column n, k-contiguous.
__device__ inline v16h frag_b_ld(const _Float16* s, int n0, int ld) {
  const int lane  = threadIdx.x & 31;
  const _Float16* p = s + (size_t)(n0 + (lane & 15)) * ld + ((lane >> 4) << 3);
  v8h lo = *(const v8h*)p;
  v8h hi = *(const v8h*)(p + 16);
  v16h f;
#pragma unroll
  for (int i = 0; i < 8; ++i) { f[i] = lo[i]; f[8 + i] = hi[i]; }
  return f;
}

__device__ inline v8f wmma_f16(v16h a, v16h b, v8f c) {
  return __builtin_amdgcn_wmma_f32_16x16x32_f16(false, a, false, b,
                                                (short)0, c, false, false);
}

// ---------------- TDM: contiguous 49x32 f16 tile -> LDS with row padding ----
// D# per CDNA5 ISA.  data_size=4B, 1 row of 784 dwords;
// pad_enable inserts 4 DWORDs (8 halves) after every 16 DWORDs (32 halves)
// -> LDS row stride of 40 halves, matching frag_a_ld / frag_b_ld.
__device__ inline void tdm_load_tile49x32(const void* gptr, unsigned lds_off) {
  unsigned long long ga = (unsigned long long)(uintptr_t)gptr;
  v4u g0;
  g0[0] = 1u;                                   // count=1, user descriptor
  g0[1] = lds_off;                              // LDS byte address
  g0[2] = (unsigned)(ga & 0xffffffffu);         // global_addr[31:0]
  g0[3] = (unsigned)((ga >> 32) & 0x01ffffffu)  // global_addr[56:32]
          | (2u << 30);                         // type = 2 ("image")
  v8i g1;
  g1[0] = (2 << 16)      // data_size = 4B
        | (1 << 20)      // pad_enable
        | (3 << 22)      // pad_interval: 16 DWORDs (one 32-half row)
        | (3 << 25);     // pad_amount:   4 DWORDs (8 halves)
  g1[1] = (int)(784u << 16);   // tensor_dim0[15:0]  (bits 63:48)
  g1[2] = (int)(1u << 16);     // tensor_dim0 hi=0 | tensor_dim1 lo = 1
  g1[3] = (int)(784u << 16);   // tensor_dim1 hi=0 | tile_dim0 = 784
  g1[4] = 1;                   // tile_dim1 = 1, tile_dim2 = 0
  g1[5] = 784;                 // tensor_dim0_stride lo
  g1[6] = 0;                   // stride hi | tensor_dim1_stride lo
  g1[7] = 0;
  v4i z4 = {0, 0, 0, 0};
#if __clang_major__ >= 23
  v8i z8 = {0, 0, 0, 0, 0, 0, 0, 0};
  __builtin_amdgcn_tensor_load_to_lds(g0, g1, z4, z4, z8, 0);
#else
  __builtin_amdgcn_tensor_load_to_lds(g0, g1, z4, z4, 0);
#endif
}

// ---------------- weight prep: fp32 row-major [K][N] -> f16 [N][K] ----------
__global__ __launch_bounds__(256) void prep_wt(const float* __restrict__ w,
                                               _Float16* __restrict__ wt,
                                               int K, int N) {
  int idx = blockIdx.x * 256 + threadIdx.x;
  if (idx < K * N) {
    int n = idx / K, k = idx - n * K;
    wt[idx] = (_Float16)w[(size_t)k * N + n];
  }
}

// ---------------- LayerNorm (optionally fused roll + window partition) ------
template<bool WINDOW>
__global__ __launch_bounds__(256) void ln_kernel(const float* __restrict__ x,
                                                 const float* __restrict__ g,
                                                 const float* __restrict__ bta,
                                                 _Float16* __restrict__ h) {
  const int lane = threadIdx.x & 31;
  const int wid  = threadIdx.x >> 5;
  const size_t t = (size_t)blockIdx.x * 8 + wid;   // destination token
  const float* src;
  if (WINDOW) {
    int b = (int)(t / 3136), r = (int)(t % 3136);
    int w = r / NTOK, n = r - w * NTOK;
    int wh = w >> 3, ww = w & 7;
    int ii = wh * WS_ + n / WS_, jj = ww * WS_ + n % WS_;
    int i2 = ii + 3; if (i2 >= HH) i2 -= HH;
    int j2 = jj + 3; if (j2 >= HH) j2 -= HH;
    src = x + ((size_t)b * 3136 + i2 * HH + j2) * DIM;
  } else {
    src = x + t * DIM;
  }
  float vals[6], s = 0.f, s2 = 0.f;
#pragma unroll
  for (int e = 0; e < 6; ++e) {
    float v = src[lane + 32 * e];
    vals[e] = v; s += v; s2 += v * v;
  }
#pragma unroll
  for (int off = 16; off > 0; off >>= 1) {
    s  += __shfl_xor(s,  off, 32);
    s2 += __shfl_xor(s2, off, 32);
  }
  const float mean = s * (1.f / DIM);
  const float var  = s2 * (1.f / DIM) - mean * mean;
  const float rstd = rsqrtf(var + 1e-5f);
#pragma unroll
  for (int e = 0; e < 6; ++e) {
    int c = lane + 32 * e;
    h[t * DIM + c] = (_Float16)((vals[e] - mean) * rstd * g[c] + bta[c]);
  }
}

// ---------------- generic f16 WMMA GEMM with fused epilogues ----------------
// C[M,N] = A[M,K] * Bt[N,K]^T + bias.  Tile 256x64, 8 waves, each wave a
// 32x64 sub-tile (2x4 WMMA): 6 fragment loads per 8 WMMAs per k-step.
enum { EPI_QKV = 0, EPI_PROJ = 1, EPI_GELU = 2, EPI_MLP2 = 3 };

template<int EPI>
__global__ __launch_bounds__(256) void gemm_f16(
    const _Float16* __restrict__ A, const _Float16* __restrict__ Bt,
    const float* __restrict__ bias, int M, int K, int N,
    _Float16* __restrict__ o_q, _Float16* __restrict__ o_k,
    _Float16* __restrict__ o_v,
    const float* __restrict__ resid, float* __restrict__ o_f) {
  constexpr int LDA = 40;                 // 32 + 8 halves (80B rows)
  constexpr int BM = 256, BN = 64;
  __shared__ alignas(16) _Float16 sA[2][BM * LDA];
  __shared__ alignas(16) _Float16 sB[2][BN * LDA];

  const int tid  = threadIdx.x;
  const int lane = tid & 31, wid = tid >> 5;       // wid == wave M block
  const int m0 = blockIdx.y * BM, n0 = blockIdx.x * BN;
  const int ak = (tid & 3) << 3;

  v8h ra[4], rb;
  auto gload = [&](int k0) {
#pragma unroll
    for (int j = 0; j < 4; ++j)
      ra[j] = *(const v8h*)(A + (size_t)(m0 + ((tid + 256 * j) >> 2)) * K +
                            k0 + ak);
    rb = *(const v8h*)(Bt + (size_t)(n0 + (tid >> 2)) * K + k0 + ak);
  };
  auto sstore = [&](int b) {
#pragma unroll
    for (int j = 0; j < 4; ++j)
      *(v8h*)(&sA[b][((tid + 256 * j) >> 2) * LDA + ak]) = ra[j];
    *(v8h*)(&sB[b][(tid >> 2) * LDA + ak]) = rb;
  };

  v8f acc[2][4] = {};
  const int ksteps = K >> 5;
  gload(0);
  sstore(0);
  __syncthreads();
  int buf = 0;
  for (int ks = 0; ks < ksteps; ++ks) {
    if (ks + 1 < ksteps) gload((ks + 1) << 5);
    if (ks + 2 < ksteps) {                         // global_prefetch_b8
      __builtin_prefetch(A  + (size_t)(m0 + (tid >> 2)) * K + ((ks + 2) << 5), 0, 0);
      __builtin_prefetch(Bt + (size_t)(n0 + (tid >> 2)) * K + ((ks + 2) << 5), 0, 0);
    }
    v16h a0 = frag_a_ld(&sA[buf][0], wid * 32,      LDA);
    v16h a1 = frag_a_ld(&sA[buf][0], wid * 32 + 16, LDA);
    v16h bf[4];
#pragma unroll
    for (int j = 0; j < 4; ++j) bf[j] = frag_b_ld(&sB[buf][0], j * 16, LDA);
#pragma unroll
    for (int j = 0; j < 4; ++j) {
      acc[0][j] = wmma_f16(a0, bf[j], acc[0][j]);
      acc[1][j] = wmma_f16(a1, bf[j], acc[1][j]);
    }
    if (ks + 1 < ksteps) { buf ^= 1; sstore(buf); }
    __syncthreads();
  }

  // epilogue: C/D layout — VGPR r: lanes 0-15 M=r, lanes 16-31 M=8+r, N=lane&15
#pragma unroll
  for (int i = 0; i < 2; ++i)
#pragma unroll
    for (int j = 0; j < 4; ++j)
#pragma unroll
      for (int r = 0; r < 8; ++r) {
        const int row = m0 + wid * 32 + i * 16 + r + ((lane >> 4) << 3);
        const int col = n0 + j * 16 + (lane & 15);
        float v = acc[i][j][r] + bias[col];
        if constexpr (EPI == EPI_QKV) {
          const int which = col / DIM;
          const int head  = (col % DIM) >> 5;
          const int d     = col & 31;
          const int w = row / NTOK, n = row - w * NTOK;
          const size_t dst = (((size_t)w * NH + head) * NTOK + n) * HD + d;
          if (which == 0)      o_q[dst] = (_Float16)(v * QSCALE);
          else if (which == 1) o_k[dst] = (_Float16)v;
          else                 o_v[dst] = (_Float16)v;
        } else if constexpr (EPI == EPI_PROJ) {
          const int b = row / 3136, r2 = row - b * 3136;
          const int w = r2 / NTOK, n = r2 - w * NTOK;
          const int wh = w >> 3, ww = w & 7;
          const int ii = wh * WS_ + n / WS_, jj = ww * WS_ + n % WS_;
          int i2 = ii + 3; if (i2 >= HH) i2 -= HH;
          int j2 = jj + 3; if (j2 >= HH) j2 -= HH;
          const size_t idx = ((size_t)b * 3136 + i2 * HH + j2) * DIM + col;
          o_f[idx] = resid[idx] + v;
        } else if constexpr (EPI == EPI_GELU) {
          const float ge = 0.5f * v * (1.0f + erff(v * 0.70710678118f));
          o_q[(size_t)row * N + col] = (_Float16)ge;
        } else {  // EPI_MLP2: final residual add onto x1 held in o_f
          const size_t idx = (size_t)row * N + col;
          o_f[idx] = o_f[idx] + v;
        }
      }
}

// ---------------- per-(window,head) attention -------------------------------
// blockIdx.x = w_global * NH + head ; 128 threads (4 waves).
// Q/K tiles arrive via the Tensor Data Mover with DMA row padding.
__global__ __launch_bounds__(128) void attn_win(
    const _Float16* __restrict__ qb, const _Float16* __restrict__ kb,
    const _Float16* __restrict__ vb, const float* __restrict__ btab,
    const int* __restrict__ ridx_g, const float* __restrict__ amask,
    _Float16* __restrict__ attn_out) {
  __shared__ alignas(16) _Float16 sQ[64 * 40];
  __shared__ alignas(16) _Float16 sK[64 * 40];
  __shared__ alignas(16) _Float16 sVt[32 * 72];   // [dim][token]
  __shared__ alignas(16) float    sS[64 * 65];    // padded rows
  __shared__ alignas(16) _Float16 sP[64 * 72];

  const int tid = threadIdx.x, lane = tid & 31, wid = tid >> 5;
  const int head = blockIdx.x % NH;
  const int w    = blockIdx.x / NH;     // global window 0..4095
  const int wimg = w & (NWIN - 1);      // window within image (mask index)

  const _Float16* q = qb + (size_t)blockIdx.x * NTOK * HD;
  const _Float16* k = kb + (size_t)blockIdx.x * NTOK * HD;
  const _Float16* v = vb + (size_t)blockIdx.x * NTOK * HD;

  if (wid == 0) {                       // one wave drives the TDM
    tdm_load_tile49x32(q, (unsigned)(uintptr_t)&sQ[0]);
    tdm_load_tile49x32(k, (unsigned)(uintptr_t)&sK[0]);
  }
  // V transposed (and zero-padded) by hand; pad must be exact zeros since
  // softmax-zeroed P columns multiply it.
  for (int idx = tid; idx < 64 * HD; idx += 128) {
    const int n = idx >> 5, d = idx & 31;
    sVt[d * 72 + n] = (n < NTOK) ? v[n * HD + d] : (_Float16)0.f;
  }
  if (wid == 0) __builtin_amdgcn_s_wait_tensorcnt(0);
  __syncthreads();

  // S = (q*scale) @ k^T : each wave one 16-row block, 4 col tiles, K=32
  const int m0 = wid * 16;
  {
    v16h aq = frag_a_ld(sQ, m0, 40);
    v8f sacc[4] = {};
#pragma unroll
    for (int j = 0; j < 4; ++j) {
      v16h bk = frag_b_ld(sK, j * 16, 40);
      sacc[j] = wmma_f16(aq, bk, sacc[j]);
    }
#pragma unroll
    for (int j = 0; j < 4; ++j)
#pragma unroll
      for (int r = 0; r < 8; ++r) {
        const int row = m0 + r + ((lane >> 4) << 3);
        sS[row * 65 + j * 16 + (lane & 15)] = sacc[j][r];
      }
  }
  __syncthreads();

  // softmax rows (rel bias + shift mask), build f16 P (zero-padded to 64x64)
  if (tid < 64) {
    const int n = tid;
    float* srow = &sS[n * 65];
    if (n < NTOK) {
      const int*   ri = ridx_g + n * NTOK;
      const float* mr = amask + ((size_t)wimg * NTOK + n) * NTOK;
      float mx = -3.0e38f;
      for (int m = 0; m < NTOK; ++m) {
        float val = srow[m] + btab[ri[m] * NH + head] + mr[m];
        srow[m] = val;
        mx = fmaxf(mx, val);
      }
      float sum = 0.f;
      for (int m = 0; m < NTOK; ++m) {
        float e = __expf(srow[m] - mx);
        srow[m] = e; sum += e;
      }
      const float inv = 1.f / sum;
      for (int m = 0; m < 64; ++m)
        sP[n * 72 + m] = (m < NTOK) ? (_Float16)(srow[m] * inv) : (_Float16)0.f;
    } else {
      for (int m = 0; m < 64; ++m) sP[n * 72 + m] = (_Float16)0.f;
    }
  }
  __syncthreads();

  // O = P @ V : K=64 (2 steps), 2 col tiles of 16 dims
  v8f oacc[2] = {};
#pragma unroll
  for (int ks = 0; ks < 2; ++ks) {
    v16h ap = frag_a_ld(sP + ks * 32, m0, 72);
#pragma unroll
    for (int j = 0; j < 2; ++j) {
      v16h bv = frag_b_ld(sVt + ks * 32, j * 16, 72);
      oacc[j] = wmma_f16(ap, bv, oacc[j]);
    }
  }
#pragma unroll
  for (int j = 0; j < 2; ++j)
#pragma unroll
    for (int r = 0; r < 8; ++r) {
      const int m = m0 + r + ((lane >> 4) << 3);
      if (m < NTOK) {
        const int d = j * 16 + (lane & 15);
        attn_out[((size_t)w * NTOK + m) * DIM + head * HD + d] =
            (_Float16)oacc[j][r];
      }
    }
}

// ---------------------------------------------------------------------------
extern "C" void kernel_launch(void* const* d_in, const int* in_sizes, int n_in,
                              void* d_out, int out_size, void* d_ws,
                              size_t ws_size, hipStream_t stream) {
  const float* x      = (const float*)d_in[0];
  const float* gamma1 = (const float*)d_in[1];
  const float* beta1  = (const float*)d_in[2];
  const float* qkv_w  = (const float*)d_in[3];
  const float* qkv_b  = (const float*)d_in[4];
  const float* rbt    = (const float*)d_in[5];
  const int*   ridx   = (const int*)  d_in[6];
  const float* proj_w = (const float*)d_in[7];
  const float* proj_b = (const float*)d_in[8];
  const float* gamma2 = (const float*)d_in[9];
  const float* beta2  = (const float*)d_in[10];
  const float* w1     = (const float*)d_in[11];
  const float* b1     = (const float*)d_in[12];
  const float* w2     = (const float*)d_in[13];
  const float* b2     = (const float*)d_in[14];
  const float* amask  = (const float*)d_in[15];
  float* out = (float*)d_out;

  char* ws = (char*)d_ws;
  constexpr size_t HALF = 2;
  constexpr size_t o_wqkv = 0;
  constexpr size_t o_wprj = o_wqkv + (size_t)576 * DIM * HALF;
  constexpr size_t o_w1t  = o_wprj + (size_t)DIM * DIM * HALF;
  constexpr size_t o_w2t  = o_w1t  + (size_t)MLPD * DIM * HALF;
  constexpr size_t o_hwin = o_w2t  + (size_t)DIM * MLPD * HALF;
  constexpr size_t szTok  = (size_t)MTOT * DIM * HALF;            // 77 MB
  constexpr size_t o_q    = o_hwin + szTok;
  constexpr size_t o_k    = o_q + szTok;
  constexpr size_t o_v    = o_k + szTok;
  constexpr size_t o_ao   = o_v + szTok;        // attn_out f16 [M][192]
  constexpr size_t o_g    = o_q;                // MLP hidden reuses q/k/v/ao
  constexpr size_t o_h2   = o_hwin;             // LN2 output reuses h_win

  _Float16* wqkv_t = (_Float16*)(ws + o_wqkv);
  _Float16* wprj_t = (_Float16*)(ws + o_wprj);
  _Float16* w1t    = (_Float16*)(ws + o_w1t);
  _Float16* w2t    = (_Float16*)(ws + o_w2t);
  _Float16* hwin   = (_Float16*)(ws + o_hwin);
  _Float16* qbuf   = (_Float16*)(ws + o_q);
  _Float16* kbuf   = (_Float16*)(ws + o_k);
  _Float16* vbuf   = (_Float16*)(ws + o_v);
  _Float16* aobuf  = (_Float16*)(ws + o_ao);
  _Float16* gbuf   = (_Float16*)(ws + o_g);
  _Float16* h2buf  = (_Float16*)(ws + o_h2);

  // 1) weights -> f16 transposed [N][K]
  prep_wt<<<(DIM * 576 + 255) / 256, 256, 0, stream>>>(qkv_w, wqkv_t, DIM, 576);
  prep_wt<<<(DIM * DIM + 255) / 256, 256, 0, stream>>>(proj_w, wprj_t, DIM, DIM);
  prep_wt<<<(DIM * MLPD + 255) / 256, 256, 0, stream>>>(w1, w1t, DIM, MLPD);
  prep_wt<<<(MLPD * DIM + 255) / 256, 256, 0, stream>>>(w2, w2t, MLPD, DIM);

  // 2) LN1 + roll + window partition -> f16 tokens (window order)
  ln_kernel<true><<<MTOT / 8, 256, 0, stream>>>(x, gamma1, beta1, hwin);

  // 3) QKV GEMM, scatter into per-head q/k/v (q pre-scaled)
  gemm_f16<EPI_QKV><<<dim3(576 / 64, MTOT / 256), 256, 0, stream>>>(
      hwin, wqkv_t, qkv_b, MTOT, DIM, 576, qbuf, kbuf, vbuf, nullptr, nullptr);

  // 4) windowed attention (4096 windows x 6 heads)
  attn_win<<<4096 * NH, 128, 0, stream>>>(qbuf, kbuf, vbuf, rbt, ridx, amask,
                                          aobuf);

  // 5) proj GEMM + window reverse + roll back + residual -> x1 (in d_out)
  gemm_f16<EPI_PROJ><<<dim3(DIM / 64, MTOT / 256), 256, 0, stream>>>(
      aobuf, wprj_t, proj_b, MTOT, DIM, DIM, nullptr, nullptr, nullptr, x, out);

  // 6) LN2 on x1 -> f16
  ln_kernel<false><<<MTOT / 8, 256, 0, stream>>>(out, gamma2, beta2, h2buf);

  // 7) MLP up + exact GELU -> f16 hidden
  gemm_f16<EPI_GELU><<<dim3(MLPD / 64, MTOT / 256), 256, 0, stream>>>(
      h2buf, w1t, b1, MTOT, DIM, MLPD, gbuf, nullptr, nullptr, nullptr, nullptr);

  // 8) MLP down + residual onto x1 -> final output
  gemm_f16<EPI_MLP2><<<dim3(DIM / 64, MTOT / 256), 256, 0, stream>>>(
      gbuf, w2t, b2, MTOT, MLPD, DIM, nullptr, nullptr, nullptr, nullptr, out);
}
